// Dense_45045617001118
// MI455X (gfx1250) — 8-bit matrix operand order repaired and checked by reading, compiled, not run on hardware
//
#include <hip/hip_runtime.h>


typedef int v8i __attribute__((ext_vector_type(8)));
typedef int v4i __attribute__((ext_vector_type(4)));

#define M_DIM 1024
#define K_DIM 2048
#define N_DIM 1024
#define AP 80
#define TP 68
#define NSTEP (K_DIM / 64)

#if __has_builtin(__builtin_amdgcn_global_load_async_to_lds_b128) && \
    __has_builtin(__builtin_amdgcn_s_wait_asynccnt)
#define USE_ASYNC_LDS 1
typedef __attribute__((address_space(1))) v4i* g_v4i_p;
typedef __attribute__((address_space(3))) v4i* l_v4i_p;
#else
#define USE_ASYNC_LDS 0
#endif

__global__ void quant_x_kernel(const float* __restrict__ x,
                               signed char* __restrict__ aq, int n4) {
  int i = blockIdx.x * blockDim.x + threadIdx.x;
  if (i < n4) {
    float4 f = ((const float4*)x)[i];
    char4 c;
    c.x = (signed char)truncf(f.x);
    c.y = (signed char)truncf(f.y);
    c.z = (signed char)truncf(f.z);
    c.w = (signed char)truncf(f.w);
    ((char4*)aq)[i] = c;
  }
}

__global__ void quant_wT_kernel(const float* __restrict__ W,
                                signed char* __restrict__ btq) {
  __shared__ signed char tile[64 * TP];
  const int k0 = blockIdx.x * 64;
  const int j0 = blockIdx.y * 64;
  const int t  = threadIdx.x;

  {
    int r  = t >> 4;
    int jc = (t & 15) * 4;
#pragma unroll
    for (int i = 0; i < 4; ++i) {
      int row = r + i * 16;
      float4 f = *(const float4*)(W + (size_t)(k0 + row) * N_DIM + j0 + jc);
      char4 c;
      c.x = (signed char)truncf(f.x * 32.0f);
      c.y = (signed char)truncf(f.y * 32.0f);
      c.z = (signed char)truncf(f.z * 32.0f);
      c.w = (signed char)truncf(f.w * 32.0f);
      *(char4*)(&tile[row * TP + jc]) = c;
    }
  }
  __syncthreads();
  {
    int j  = t >> 2;
    int kc = (t & 3) * 16;
    union { int4 v; signed char c[16]; } u;
#pragma unroll
    for (int q = 0; q < 16; ++q) u.c[q] = tile[(kc + q) * TP + j];
    *(int4*)(btq + (size_t)(j0 + j) * K_DIM + k0 + kc) = u.v;
  }
}

__global__ __launch_bounds__(128, 1)
void gemm_wmma_iu8_kernel(const signed char* __restrict__ Aq,
                          const signed char* __restrict__ BTq,
                          const float* __restrict__ bias,
                          float* __restrict__ out) {
  __shared__ signed char lA[2][64 * AP];
  __shared__ signed char lB[2][64 * AP];

  const int m_blk = blockIdx.y * 64;
  const int n_blk = blockIdx.x * 64;
  const int tid  = threadIdx.x;
  const int lane = tid & 31;
  const int wave = tid >> 5;
  const int wm   = wave >> 1;
  const int wn   = wave & 1;
  const int half = lane >> 4;
  const int l15  = lane & 15;

  auto load_tile = [&](int k0, int buf) {
#pragma unroll
    for (int s = 0; s < 2; ++s) {
      int c    = tid + s * 128;
      int row  = c >> 2;
      int koff = (c & 3) * 16;
      const signed char* ga = Aq + (size_t)(m_blk + row) * K_DIM + k0 + koff;
      const signed char* gb = BTq + (size_t)(n_blk + row) * K_DIM + k0 + koff;
#if USE_ASYNC_LDS
      __builtin_amdgcn_global_load_async_to_lds_b128(
          (g_v4i_p)(void*)ga, (l_v4i_p)(void*)&lA[buf][row * AP + koff], 0, 0);
      __builtin_amdgcn_global_load_async_to_lds_b128(
          (g_v4i_p)(void*)gb, (l_v4i_p)(void*)&lB[buf][row * AP + koff], 0, 0);
#else
      int4 va = *(const int4*)ga;
      int4 vb = *(const int4*)gb;
      *(int4*)(&lA[buf][row * AP + koff]) = va;
      *(int4*)(&lB[buf][row * AP + koff]) = vb;
#endif
    }
  };

  v8i acc[2][2] = {};

  load_tile(0, 0);

  for (int i = 0; i < NSTEP; ++i) {
    const int cur = i & 1;
#if USE_ASYNC_LDS
    __builtin_amdgcn_s_wait_asynccnt(0);
#endif
    __syncthreads();
    if (i + 1 < NSTEP) load_tile((i + 1) * 64, 1 ^ cur);

    v8i af[2], bf[2];
#pragma unroll
    for (int f = 0; f < 2; ++f) {
      int row = wm * 32 + f * 16 + l15;
      int kh  = half * 8;
      union { v8i v; long l[4]; } u;
#pragma unroll
      for (int c2 = 0; c2 < 4; ++c2)
        u.l[c2] = *(const long*)(&lA[cur][row * AP + (c2 >> 1) * 32 + kh * 2 + (c2 & 1) * 8]);
      af[f] = u.v;
    }
#pragma unroll
    for (int g = 0; g < 2; ++g) {
      int col = wn * 32 + g * 16 + l15;
      int kq  = half * 16;
      union { v8i v; int4 q[2]; } u;
      u.q[0] = *(const int4*)(&lB[cur][col * AP + kq]);
      u.q[1] = *(const int4*)(&lB[cur][col * AP + 32 + kq]);
      bf[g] = u.v;
    }

#pragma unroll
    for (int f = 0; f < 2; ++f)
#pragma unroll
      for (int g = 0; g < 2; ++g)
        acc[f][g] = __builtin_amdgcn_wmma_i32_16x16x64_iu8(
             true, af[f],  true, bf[g], acc[f][g],
             false,  false);
  }

#pragma unroll
  for (int f = 0; f < 2; ++f) {
#pragma unroll
    for (int g = 0; g < 2; ++g) {
      int col  = n_blk + wn * 32 + g * 16 + l15;
      float bv = bias[col];
#pragma unroll
      for (int r = 0; r < 8; ++r) {
        int row    = m_blk + wm * 32 + f * 16 + half * 8 + r;
        float accf = (float)acc[f][g][r] * (1.0f / 32.0f);
        float y    = (accf + bv) * 32.0f;
        out[(size_t)row * N_DIM + col] = truncf(y) * (1.0f / 32.0f);
      }
    }
  }
}

extern "C" void kernel_launch(void* const* d_in, const int* in_sizes, int n_in,
                              void* d_out, int out_size, void* d_ws, size_t ws_size,
                              hipStream_t stream) {
  const float* x = (const float*)d_in[0];
  const float* W = (const float*)d_in[1];
  const float* b = (const float*)d_in[2];
  float* out = (float*)d_out;

  signed char* Aq  = (signed char*)d_ws;
  signed char* BTq = Aq + (size_t)M_DIM * K_DIM;

  int n4 = (M_DIM * K_DIM) / 4;
  quant_x_kernel<<<(n4 + 255) / 256, 256, 0, stream>>>(x, Aq, n4);
  quant_wT_kernel<<<dim3(K_DIM / 64, N_DIM / 64), 256, 0, stream>>>(W, BTq);
  gemm_wmma_iu8_kernel<<<dim3(N_DIM / 64, M_DIM / 64), 128, 0, stream>>>(
      Aq, BTq, b, out);
}
